// AtomicMultiHeadAttention_49417893708042
// MI455X (gfx1250) — matrix operand order repaired and checked by reading, not compiled, not run on hardware
//
#include <hip/hip_runtime.h>

// ---------------------------------------------------------------------------
// MHA for MI455X (gfx1250): bf16 WMMA GEMMs + double-buffered flash attention.
// B=4, S=1024, D_MODEL=1024, H=16, DEPTH=64.
// ---------------------------------------------------------------------------

typedef __attribute__((ext_vector_type(16))) __bf16          v16bf;
typedef __attribute__((ext_vector_type(8)))  float           v8f;
typedef __attribute__((ext_vector_type(8)))  unsigned short  v8us;

// Async global->LDS copies (CDNA5): use when the toolchain exposes them.
// Param type (from hipcc diagnostic): pointer to int4 (b128 payload),
// global source in AS1, LDS destination in AS3.
#if __has_builtin(__builtin_amdgcn_global_load_async_to_lds_b128) && \
    __has_builtin(__builtin_amdgcn_s_wait_asynccnt)
#define USE_ASYNC 1
typedef int v4i_ __attribute__((vector_size(4 * sizeof(int))));
typedef __attribute__((address_space(1))) v4i_ gv4;   // global int4
typedef __attribute__((address_space(3))) v4i_ lv4;   // LDS int4
#else
#define USE_ASYNC 0
#endif

__device__ __forceinline__ unsigned short f2bf(float f) {
    return __builtin_bit_cast(unsigned short, (__bf16)f);   // native v_cvt, RNE
}

// Build a v16bf WMMA fragment from two 16-byte LDS/global regions.
__device__ __forceinline__ v16bf load_frag(const unsigned short* p0,
                                           const unsigned short* p1) {
    union { v16bf f; v8us h[2]; } u;
    u.h[0] = *(const v8us*)p0;
    u.h[1] = *(const v8us*)p1;
    return u.f;
}

__device__ __forceinline__ float redmax16(float v) {
#pragma unroll
    for (int d = 1; d < 16; d <<= 1) v = fmaxf(v, __shfl_xor(v, d, 32));
    return v;
}

// ---------------------------------------------------------------------------
// GEMM: C(4096,1024) = A(4096,1024) x W(1024,1024) + bias
// MODE 0: store bf16, head-split (B,H,S,64)      (Q, K projections)
// MODE 1: store bf16, transposed  (B,H,64,S)     (V projection)
// MODE 2: store fp32, plain (M,N)                (final output projection)
// ABF:    A is bf16 (true; staged via async-to-LDS) or fp32 (false; convert)
// Tile 128x128x32, 256 threads = 8 waves, each wave = 32x64 (2x4 WMMA tiles).
// Double-buffered LDS, one barrier per K step.
// ---------------------------------------------------------------------------
template <int MODE, bool ABF>
__global__ __launch_bounds__(256)
void gemm128(const void* __restrict__ Ap, const float* __restrict__ W,
             const float* __restrict__ bias, void* __restrict__ Out) {
    constexpr int Kdim = 1024, Ndim = 1024;
    __shared__ __align__(16) unsigned short At[2][128][32];   // [m][k] bf16
    __shared__ __align__(16) unsigned short Bt[2][128][32];   // [n][k] bf16 (transposed)

    const int tid  = threadIdx.x;
    const int lane = tid & 31, wave = tid >> 5;
    const int bm   = blockIdx.y * 128, bn = blockIdx.x * 128;
    const int wm   = (wave & 3) * 32, wn = (wave >> 2) * 64;
    const int lr   = lane & 15, kh = lane >> 4;

    v8f zero = {};
    v8f acc[2][4];
#pragma unroll
    for (int i = 0; i < 2; ++i)
#pragma unroll
        for (int j = 0; j < 4; ++j) acc[i][j] = zero;

    const int arow = tid >> 1, akq = (tid & 1) * 16;   // A copy: 128 rows x 16 k each
    const int brow = tid >> 3, bng = (tid & 7) * 16;   // W copy: 32 k rows x 16 n each

    auto stage = [&](int k0, int buf) {
        // ---- stage A tile ----
        if constexpr (ABF) {
            const unsigned short* Ab =
                (const unsigned short*)Ap + (size_t)(bm + arow) * Kdim + k0 + akq;
#if USE_ASYNC
            __builtin_amdgcn_global_load_async_to_lds_b128(
                (gv4*)Ab, (lv4*)&At[buf][arow][akq], 0, 0);
            __builtin_amdgcn_global_load_async_to_lds_b128(
                (gv4*)(Ab + 8), (lv4*)&At[buf][arow][akq + 8], 0, 0);
#else
            *(v8us*)&At[buf][arow][akq]     = *(const v8us*)Ab;
            *(v8us*)&At[buf][arow][akq + 8] = *(const v8us*)(Ab + 8);
#endif
        } else {
            const float* Af = (const float*)Ap + (size_t)(bm + arow) * Kdim + k0 + akq;
            unsigned short t[16] __attribute__((aligned(16)));
#pragma unroll
            for (int g = 0; g < 4; ++g) {
                float4 f = *(const float4*)(Af + g * 4);
                t[g*4+0] = f2bf(f.x); t[g*4+1] = f2bf(f.y);
                t[g*4+2] = f2bf(f.z); t[g*4+3] = f2bf(f.w);
            }
            *(v8us*)&At[buf][arow][akq]     = *(const v8us*)t;
            *(v8us*)&At[buf][arow][akq + 8] = *(const v8us*)(t + 8);
            if (k0 + 32 < Kdim)
                __builtin_prefetch(Af + 32, 0, 1);    // -> global_prefetch_b8
        }
        // ---- stage W tile transposed: Bt[n][k] (fp32 -> bf16) ----
        {
            const float* Wf = W + (size_t)(k0 + brow) * Ndim + bn + bng;
#pragma unroll
            for (int g = 0; g < 4; ++g) {
                float4 f = *(const float4*)(Wf + g * 4);
                Bt[buf][bng + g*4 + 0][brow] = f2bf(f.x);
                Bt[buf][bng + g*4 + 1][brow] = f2bf(f.y);
                Bt[buf][bng + g*4 + 2][brow] = f2bf(f.z);
                Bt[buf][bng + g*4 + 3][brow] = f2bf(f.w);
            }
            if (k0 + 32 < Kdim)
                __builtin_prefetch(Wf + 32 * Ndim, 0, 1);
        }
    };

    stage(0, 0);
#if USE_ASYNC
    if constexpr (ABF) __builtin_amdgcn_s_wait_asynccnt(0);
#endif
    __syncthreads();
    int buf = 0;
    for (int k0 = 0; k0 < Kdim; k0 += 32) {
        if (k0 + 32 < Kdim) stage(k0 + 32, buf ^ 1);   // overlap with WMMAs below

        v16bf af[2];
#pragma unroll
        for (int mt = 0; mt < 2; ++mt) {
            const unsigned short* p = &At[buf][wm + mt*16 + lr][8 * kh];
            af[mt] = load_frag(p, p + 16);     // K 0-7/16-23 (half0) or 8-15/24-31
        }
#pragma unroll
        for (int nt = 0; nt < 4; ++nt) {
            const unsigned short* p = &Bt[buf][wn + nt*16 + lr][8 * kh];
            v16bf bfv = load_frag(p, p + 16);  // K 0-7/16-23 (half0) or 8-15/24-31
#pragma unroll
            for (int mt = 0; mt < 2; ++mt)
                acc[mt][nt] = __builtin_amdgcn_wmma_f32_16x16x32_bf16(
                    false, af[mt], false, bfv, (short)0, acc[mt][nt], false, false);
        }
#if USE_ASYNC
        if constexpr (ABF) __builtin_amdgcn_s_wait_asynccnt(0);
#endif
        __syncthreads();
        buf ^= 1;
    }

    // ---- epilogue: bias + store per MODE ----
#pragma unroll
    for (int nt = 0; nt < 4; ++nt) {
        const int n  = bn + wn + nt*16 + lr;
        const float bv = bias[n];
#pragma unroll
        for (int mt = 0; mt < 2; ++mt) {
#pragma unroll
            for (int r = 0; r < 8; ++r) {
                const int m = bm + wm + mt*16 + r + 8*kh;
                const float val = acc[mt][nt][r] + bv;
                if constexpr (MODE == 2) {
                    ((float*)Out)[(size_t)m * Ndim + n] = val;
                } else {
                    const int b = m >> 10, s = m & 1023;
                    const int h = n >> 6,  d = n & 63;
                    const unsigned short bw = f2bf(val);
                    if constexpr (MODE == 0)   // (B,H,S,64)
                        ((unsigned short*)Out)[(((size_t)(b*16 + h))*1024 + s)*64 + d] = bw;
                    else                       // (B,H,64,S)
                        ((unsigned short*)Out)[(((size_t)(b*16 + h))*64 + d)*1024 + s] = bw;
                }
            }
        }
    }
}

// ---------------------------------------------------------------------------
// Flash attention: one workgroup per (b, h, 128-row q block); 8 waves x 16 q
// rows. Double-buffered 32-key blocks staged via async global->LDS copies;
// log2-domain online softmax (v_exp_f32 co-executes with WMMA); row sums on
// the matrix pipe via a 5th "ones-column" WMMA accumulator.
// ---------------------------------------------------------------------------
__global__ __launch_bounds__(256)
void flash_attn(const unsigned short* __restrict__ Qh,   // (B,H,S,64) bf16
                const unsigned short* __restrict__ Kh,   // (B,H,S,64) bf16
                const unsigned short* __restrict__ Vt,   // (B,H,64,S) bf16
                const float* __restrict__ Wgt,           // (B,S,S) fp32
                const float* __restrict__ Mask,          // (B,S)   fp32
                unsigned short* __restrict__ AttnOut) {  // (B*S, 1024) bf16
    __shared__ __align__(16) unsigned short Kt[2][32][64];      // [key][depth]
    __shared__ __align__(16) unsigned short Vs[2][64][32];      // [depth][key]
    __shared__ __align__(16) float          Wt[2][128][32];     // [q_local][key_local]
    __shared__ __align__(16) unsigned short Pt[8][16][32];      // per-wave P tile

    const int tid = threadIdx.x, lane = tid & 31, wave = tid >> 5;
    const int lr = lane & 15, kh2 = lane >> 4;
    const int q0 = blockIdx.x * 128;
    const int h  = blockIdx.y, b = blockIdx.z;
    const size_t bh = (size_t)(b * 16 + h);

    // log2-domain scales: logits_log2 = qk*(0.125*log2e) + w*(scale_h*log2e)
    //                                   + mask*(-1e9*log2e)
    constexpr float LOG2E = 1.4426950408889634f;
    const float qsc = 0.125f * LOG2E;
    const float wsc = (0.1f + (float)h * (9.9f / 15.0f)) * LOG2E;
    constexpr float msc = -1e9f * LOG2E;

    // Q fragments for this wave's 16 rows (depth 64 -> two K=32 steps), in VGPRs.
    v16bf qf[2];
    {
        const unsigned short* qrow = Qh + (bh * 1024 + q0 + wave*16 + lr) * 64;
#pragma unroll
        for (int t = 0; t < 2; ++t)
            qf[t] = load_frag(qrow + 32*t + 8*kh2, qrow + 32*t + 16 + 8*kh2);
    }

    // B-fragment whose column 0 is all-ones: accumulates row sums of P via WMMA.
    v16bf onesf;
    {
        union { v16bf f; unsigned short u[16]; } t;
        const unsigned short val = (lr == 0) ? (unsigned short)0x3F80 : (unsigned short)0;
#pragma unroll
        for (int i = 0; i < 16; ++i) t.u[i] = val;
        onesf = t.f;
    }

    float mx[8];
#pragma unroll
    for (int r = 0; r < 8; ++r) mx[r] = -1e30f;
    v8f zero = {};
    v8f o[5];                                   // o[0..3] = O, o[4] = row sums
#pragma unroll
    for (int nt = 0; nt < 5; ++nt) o[nt] = zero;

    const float* wbase = Wgt + (size_t)b * 1024 * 1024 + (size_t)q0 * 1024;
    const float* mrow  = Mask + (size_t)b * 1024;

    const int krow = tid >> 3, kc8 = (tid & 7) * 8;   // K-tile copy: 32x64 bf16
    const int vdep = tid >> 2, vc8 = (tid & 3) * 8;   // V-tile copy: 64x32 bf16
    const int wql  = tid >> 1, wcg = (tid & 1) * 16;  // weight copy: 128x32 fp32

    auto stage = [&](int kb, int buf) {
        const unsigned short* gk = Kh + (bh*1024 + kb + krow)*64 + kc8;
        const unsigned short* gv = Vt + (bh*64 + vdep)*1024 + kb + vc8;
        const float*          gw = wbase + (size_t)wql * 1024 + kb + wcg;
#if USE_ASYNC
        __builtin_amdgcn_global_load_async_to_lds_b128(
            (gv4*)gk, (lv4*)&Kt[buf][krow][kc8], 0, 0);
        __builtin_amdgcn_global_load_async_to_lds_b128(
            (gv4*)gv, (lv4*)&Vs[buf][vdep][vc8], 0, 0);
#pragma unroll
        for (int g = 0; g < 4; ++g)
            __builtin_amdgcn_global_load_async_to_lds_b128(
                (gv4*)(gw + g*4), (lv4*)&Wt[buf][wql][wcg + g*4], 0, 0);
#else
        *(v8us*)&Kt[buf][krow][kc8] = *(const v8us*)gk;
        *(v8us*)&Vs[buf][vdep][vc8] = *(const v8us*)gv;
#pragma unroll
        for (int g = 0; g < 4; ++g)
            *(float4*)&Wt[buf][wql][wcg + g*4] = *(const float4*)(gw + g*4);
        if (kb + 32 < 1024) {   // prefetch next block (-> global_prefetch_b8)
            __builtin_prefetch(gk + 32 * 64, 0, 1);
            __builtin_prefetch(gv + 32, 0, 1);
            __builtin_prefetch(gw + 32, 0, 1);
        }
#endif
    };

    stage(0, 0);
#if USE_ASYNC
    __builtin_amdgcn_s_wait_asynccnt(0);
#endif
    __syncthreads();
    int buf = 0;
    for (int kb = 0; kb < 1024; kb += 32) {
        if (kb + 32 < 1024) stage(kb + 32, buf ^ 1);   // overlap with compute

        // ---- S = Q K^T over this 32-key block (two 16-col accumulators) ----
        v8f s0 = zero, s1 = zero;
#pragma unroll
        for (int t = 0; t < 2; ++t) {
            const unsigned short* p0 = &Kt[buf][lr][32*t + 8*kh2];
            v16bf kf0 = load_frag(p0, p0 + 16);
            s0 = __builtin_amdgcn_wmma_f32_16x16x32_bf16(
                false, qf[t], false, kf0, (short)0, s0, false, false);
            const unsigned short* p1 = &Kt[buf][16 + lr][32*t + 8*kh2];
            v16bf kf1 = load_frag(p1, p1 + 16);
            s1 = __builtin_amdgcn_wmma_f32_16x16x32_bf16(
                false, qf[t], false, kf1, (short)0, s1, false, false);
        }

        const float mv0 = mrow[kb + lr]      * msc;
        const float mv1 = mrow[kb + 16 + lr] * msc;

        // ---- fused bias + log2-domain online softmax ----
#pragma unroll
        for (int r = 0; r < 8; ++r) {
            const int m = r + 8*kh2;
            const float a0 = s0[r]*qsc + Wt[buf][wave*16 + m][lr]      * wsc + mv0;
            const float a1 = s1[r]*qsc + Wt[buf][wave*16 + m][16 + lr] * wsc + mv1;
            float nm = redmax16(fmaxf(a0, a1));
            nm = fmaxf(nm, mx[r]);
            const float c = __builtin_amdgcn_exp2f(mx[r] - nm);
            mx[r] = nm;
            const float p0v = __builtin_amdgcn_exp2f(a0 - nm);
            const float p1v = __builtin_amdgcn_exp2f(a1 - nm);
#pragma unroll
            for (int nt = 0; nt < 5; ++nt) o[nt][r] *= c;   // rescale O and sums
            Pt[wave][m][lr]      = f2bf(p0v);     // restage P: C-layout -> LDS
            Pt[wave][m][16 + lr] = f2bf(p1v);
        }

        // ---- O += P V, sums += P ones : A-frag from Pt ----
        const unsigned short* pp = &Pt[wave][lr][8 * kh2];
        v16bf pf = load_frag(pp, pp + 16);
#pragma unroll
        for (int nt = 0; nt < 4; ++nt) {
            const unsigned short* vp = &Vs[buf][nt*16 + lr][8 * kh2];
            v16bf vf = load_frag(vp, vp + 16);
            o[nt] = __builtin_amdgcn_wmma_f32_16x16x32_bf16(
                false, pf, false, vf, (short)0, o[nt], false, false);
        }
        o[4] = __builtin_amdgcn_wmma_f32_16x16x32_bf16(
            false, pf, false, onesf, (short)0, o[4], false, false);

#if USE_ASYNC
        __builtin_amdgcn_s_wait_asynccnt(0);
#endif
        __syncthreads();
        buf ^= 1;
    }

    // ---- normalize (broadcast row sum from column-0 lane) and store bf16 ----
#pragma unroll
    for (int r = 0; r < 8; ++r) {
        const int m = r + 8*kh2;
        const float sum = __shfl(o[4][r], lane & 16, 32);   // lane 0 / lane 16
        const float inv = 1.0f / sum;
        const size_t row = (size_t)(b*1024 + q0 + wave*16 + m);
#pragma unroll
        for (int nt = 0; nt < 4; ++nt)
            AttnOut[row*1024 + h*64 + nt*16 + lr] = f2bf(o[nt][r] * inv);
    }
}

// ---------------------------------------------------------------------------
extern "C" void kernel_launch(void* const* d_in, const int* in_sizes, int n_in,
                              void* d_out, int out_size, void* d_ws, size_t ws_size,
                              hipStream_t stream) {
    (void)in_sizes; (void)n_in; (void)out_size; (void)ws_size;
    const float* v   = (const float*)d_in[0];
    const float* k   = (const float*)d_in[1];
    const float* q   = (const float*)d_in[2];
    const float* wgt = (const float*)d_in[3];
    const float* msk = (const float*)d_in[4];
    const float* Wq  = (const float*)d_in[5];
    const float* bq  = (const float*)d_in[6];
    const float* Wk  = (const float*)d_in[7];
    const float* bk  = (const float*)d_in[8];
    const float* Wv  = (const float*)d_in[9];
    const float* bv  = (const float*)d_in[10];
    const float* Wo  = (const float*)d_in[11];
    const float* bo  = (const float*)d_in[12];

    const size_t HSZ = (size_t)4 * 16 * 1024 * 64;   // elements per bf16 tensor
    unsigned short* qh = (unsigned short*)d_ws;      // (B,H,S,64) bf16
    unsigned short* kh = qh + HSZ;                   // (B,H,S,64) bf16
    unsigned short* vT = kh + HSZ;                   // (B,H,64,S) bf16
    unsigned short* at = vT + HSZ;                   // (B*S,1024) bf16

    dim3 gg(8, 32), blk(256);
    gemm128<0, false><<<gg, blk, 0, stream>>>(q, Wq, bq, qh);
    gemm128<0, false><<<gg, blk, 0, stream>>>(k, Wk, bk, kh);
    gemm128<1, false><<<gg, blk, 0, stream>>>(v, Wv, bv, vT);
    flash_attn<<<dim3(8, 16, 4), blk, 0, stream>>>(qh, kh, vT, wgt, msk, at);
    gemm128<2, true><<<gg, blk, 0, stream>>>(at, Wo, bo, (float*)d_out);
}
